// LocalRCT_58832462021094
// MI455X (gfx1250) — compile-verified, matrix operand order checked by reading, not run on hardware
//
#include <hip/hip_runtime.h>
#include <hip/hip_bf16.h>

// ---------------------------------------------------------------------------
// Types for CDNA5 WMMA (wave32)
// ---------------------------------------------------------------------------
typedef __attribute__((ext_vector_type(16))) __bf16 v16bf;
typedef __attribute__((ext_vector_type(8)))  float  v8f;
typedef __attribute__((ext_vector_type(4)))  float  v4f;

static __device__ __forceinline__ __bf16 f2bf(float f) {
    return (__bf16)f;    // lowers to v_cvt_pk_bf16_f32 when packed
}

#define RF      64
#define HH      512
#define WW      512
#define BB      4
#define MH      32
#define MW      32

// ---------------------------------------------------------------------------
// Generic 3x3 SAME conv on 17x17 maps, optional BN+SiLU epilogue.
// ---------------------------------------------------------------------------
__global__ void conv3x3_17_kernel(const float* __restrict__ in,
                                  const float* __restrict__ w,
                                  const float* __restrict__ bias,
                                  const float* __restrict__ gamma,
                                  const float* __restrict__ beta,
                                  const float* __restrict__ mean,
                                  const float* __restrict__ var,
                                  float* __restrict__ out,
                                  int Cin, int Cout, int bnSilu, int total)
{
    int idx = blockIdx.x * blockDim.x + threadIdx.x;
    if (idx >= total) return;
    int ow = idx % 17;
    int t  = idx / 17;
    int oh = t % 17;   t /= 17;
    int co = t % Cout;
    int b  = t / Cout;

    const float* wp = w  + (size_t)co * Cin * 9;
    const float* ip = in + (size_t)b  * Cin * 289;

    float acc = 0.f;
    for (int ci = 0; ci < Cin; ++ci) {
        const float* ic = ip + (size_t)ci * 289;
        const float* wc = wp + ci * 9;
#pragma unroll
        for (int kh = 0; kh < 3; ++kh) {
            int ih = oh + kh - 1;
            if (ih < 0 || ih > 16) continue;
#pragma unroll
            for (int kw = 0; kw < 3; ++kw) {
                int iw = ow + kw - 1;
                if (iw < 0 || iw > 16) continue;
                acc = fmaf(ic[ih * 17 + iw], wc[kh * 3 + kw], acc);
            }
        }
    }
    acc += bias[co];
    if (bnSilu) {
        float sc = gamma[co] * rsqrtf(var[co] + 1e-5f);
        float v  = (acc - mean[co]) * sc + beta[co];
        acc = v / (1.f + __expf(-v));            // SiLU
    }
    out[idx] = acc;
}

// ---------------------------------------------------------------------------
// Main attention kernel: one workgroup per (batch, tile(x,y)).
//   grid = (256, 4), block = 256 threads (8 wave32)
//   QK^T (8 WMMA) and softmax*V (2 WMMA, c padded to 16) per 16-pixel block.
//   1/sqrt(RF) folded into RK; exp/max/sum fully register+shfl resident.
// ---------------------------------------------------------------------------
__global__ __launch_bounds__(256)
void local_rct_attn_kernel(const float* __restrict__ feature,
                           const float* __restrict__ r_l,   // [B,1024,17,17]
                           const float* __restrict__ t_l,   // [B,48,17,17]
                           float* __restrict__ out)         // [B,3,512,512]
{
    const int tile = blockIdx.x;
    const int b    = blockIdx.y;
    const int tx   = tile >> 4;
    const int ty   = tile & 15;

    const int tid  = threadIdx.x;
    const int wave = tid >> 5;
    const int lane = tid & 31;
    const int lane15  = lane & 15;
    const int hiQuad  = (lane >> 4) & 1;

    __shared__ __bf16 sRK[64 * 64];                       // pre-scaled keys [r][k]
    __shared__ float  sTK[64 * 3];                        // values [k][c]
    __shared__ __align__(16) float sAtt[8][16][68];       // per-wave logits
    __shared__ __align__(16) float sS[8][16];             // per-wave 1/sum

    // ---- gather the 4-corner key / value matrices for this tile ----
    for (int idx = tid; idx < 64 * 64; idx += 256) {
        int r = idx >> 6, k = idx & 63;
        int corner = k >> 4, n = k & 15;
        int dx = corner >> 1, dy = corner & 1;
        float v = r_l[(((size_t)b * 1024 + (r * 16 + n)) * 17 + (tx + dx)) * 17 + (ty + dy)];
        sRK[r * 64 + k] = f2bf(v * 0.125f);               // fold 1/sqrt(64)
    }
    for (int idx = tid; idx < 3 * 64; idx += 256) {
        int k = idx / 3, c = idx % 3;
        int corner = k >> 4, n = k & 15;
        int dx = corner >> 1, dy = corner & 1;
        sTK[k * 3 + c] =
            t_l[(((size_t)b * 48 + (c * 16 + n)) * 17 + (tx + dx)) * 17 + (ty + dy)];
    }
    __syncthreads();

    union Frag { v16bf v; __bf16 u[16]; };

    // ---- B fragments for QK (RK: 4 k-tiles x 2 r-halves) ----
    Frag Bf[4][2];
#pragma unroll
    for (int kt = 0; kt < 4; ++kt) {
#pragma unroll
        for (int h = 0; h < 2; ++h) {
            int N = kt * 16 + lane15;
#pragma unroll
            for (int e = 0; e < 16; ++e) {
                int K = (e & 7) + ((e >> 3) << 4) + (hiQuad << 3) + (h << 5);
                Bf[kt][h].u[e] = sRK[K * 64 + N];
            }
        }
    }
    // ---- B fragments for V (TK: [64k x 16c], only c<3 non-zero) ----
    Frag Bt[2];
#pragma unroll
    for (int h = 0; h < 2; ++h) {
#pragma unroll
        for (int e = 0; e < 16; ++e) {
            int K = (e & 7) + ((e >> 3) << 4) + (hiQuad << 3) + (h << 5);
            Bt[h].u[e] = (lane15 < 3) ? f2bf(sTK[K * 3 + lane15]) : (__bf16)0.0f;
        }
    }

    // ---- loop over this wave's 8 row-blocks of 16 pixels ----
    for (int i = 0; i < 8; ++i) {
        const int rb = wave + i * 8;
        const int p0 = rb * 16;

        // A fragments: 16 pixels x 64 channels (non-temporal streaming loads)
        {
            const int p  = p0 + lane15;
            const int gx = tx * MH + (p >> 5);
            const int gw = ty * MW + (p & 31);
            const float* fbase = feature + (((size_t)b * RF) * HH + gx) * WW + gw;

            Frag A0, A1;
#pragma unroll
            for (int e = 0; e < 16; ++e) {
                int K = (e & 7) + ((e >> 3) << 4) + (hiQuad << 3);
                A0.u[e] = f2bf(__builtin_nontemporal_load(
                                   fbase + (size_t)K        * (HH * WW)));
                A1.u[e] = f2bf(__builtin_nontemporal_load(
                                   fbase + (size_t)(K + 32) * (HH * WW)));
            }

            // logits D[16p,64k] -> LDS (pre-scaled RK, no epilogue)
#pragma unroll
            for (int kt = 0; kt < 4; ++kt) {
                v8f acc = {};
                acc = __builtin_amdgcn_wmma_f32_16x16x32_bf16(
                          false, A0.v, false, Bf[kt][0].v, (short)0, acc, false, false);
                acc = __builtin_amdgcn_wmma_f32_16x16x32_bf16(
                          false, A1.v, false, Bf[kt][1].v, (short)0, acc, false, false);
#pragma unroll
                for (int v = 0; v < 8; ++v) {
                    sAtt[wave][v + (hiQuad << 3)][kt * 16 + lane15] = acc[v];
                }
            }
        }

        // ---- softmax in A-fragment layout (all 32 lanes) ----
        // lane l owns row m=l&15, k-runs {8q..8q+7} U {16+8q..} (+32 for h=1)
        const float* rowp = &sAtt[wave][lane15][0];
        float wv[32];
#pragma unroll
        for (int h = 0; h < 2; ++h) {
            v4f a = *(const v4f*)(rowp + (hiQuad << 3)      + (h << 5));
            v4f c = *(const v4f*)(rowp + (hiQuad << 3) + 4  + (h << 5));
            v4f d = *(const v4f*)(rowp + (hiQuad << 3) + 16 + (h << 5));
            v4f g = *(const v4f*)(rowp + (hiQuad << 3) + 20 + (h << 5));
#pragma unroll
            for (int j = 0; j < 4; ++j) {
                wv[h * 16 + j]      = a[j];
                wv[h * 16 + 4 + j]  = c[j];
                wv[h * 16 + 8 + j]  = d[j];
                wv[h * 16 + 12 + j] = g[j];
            }
        }
        float mx = wv[0];
#pragma unroll
        for (int e = 1; e < 32; ++e) mx = fmaxf(mx, wv[e]);
        mx = fmaxf(mx, __shfl_xor(mx, 16));       // combine row halves

        float s = 0.f;
#pragma unroll
        for (int e = 0; e < 32; ++e) {
            wv[e] = __expf(wv[e] - mx);
            s += wv[e];
        }
        s += __shfl_xor(s, 16);
        if (lane < 16) sS[wave][lane15] = 1.f / s;

        // exp weights are already in A-fragment element order
        Frag W0, W1;
#pragma unroll
        for (int e = 0; e < 16; ++e) {
            W0.u[e] = f2bf(wv[e]);
            W1.u[e] = f2bf(wv[16 + e]);
        }

        // V contraction: D2[16p,16c] = W[16p,64k] x TK[64k,16c]
        v8f acc2 = {};
        acc2 = __builtin_amdgcn_wmma_f32_16x16x32_bf16(
                   false, W0.v, false, Bt[0].v, (short)0, acc2, false, false);
        acc2 = __builtin_amdgcn_wmma_f32_16x16x32_bf16(
                   false, W1.v, false, Bt[1].v, (short)0, acc2, false, false);

        // ---- normalize + store: lane holds col c=lane15, rows v+8*hiQuad ----
        if (lane15 < 3) {
            v4f r0 = *(const v4f*)&sS[wave][(hiQuad << 3)];
            v4f r1 = *(const v4f*)&sS[wave][(hiQuad << 3) + 4];

            const int pbase = p0 + (hiQuad << 3);         // 8 consecutive pixels
            const int gx    = tx * MH + (pbase >> 5);
            const int gw    = ty * MW + (pbase & 31);
            float* op = &out[(((size_t)b * 3 + lane15) * HH + gx) * WW + gw];

            v4f o0, o1;
#pragma unroll
            for (int j = 0; j < 4; ++j) {
                o0[j] = acc2[j]     * r0[j];
                o1[j] = acc2[4 + j] * r1[j];
            }
            __builtin_nontemporal_store(o0, (v4f*)op);
            __builtin_nontemporal_store(o1, (v4f*)(op + 4));
        }
    }
}

// ---------------------------------------------------------------------------
// Launcher
// ---------------------------------------------------------------------------
extern "C" void kernel_launch(void* const* d_in, const int* in_sizes, int n_in,
                              void* d_out, int out_size, void* d_ws, size_t ws_size,
                              hipStream_t stream)
{
    const float* feature = (const float*)d_in[0];
    const float* p_low   = (const float*)d_in[1];
    const float* r_w1    = (const float*)d_in[2];
    const float* r_b1    = (const float*)d_in[3];
    const float* r_gamma = (const float*)d_in[4];
    const float* r_beta  = (const float*)d_in[5];
    const float* r_mean  = (const float*)d_in[6];
    const float* r_var   = (const float*)d_in[7];
    const float* r_w2    = (const float*)d_in[8];
    const float* r_b2    = (const float*)d_in[9];
    const float* t_w1    = (const float*)d_in[10];
    const float* t_b1    = (const float*)d_in[11];
    const float* t_gamma = (const float*)d_in[12];
    const float* t_beta  = (const float*)d_in[13];
    const float* t_mean  = (const float*)d_in[14];
    const float* t_var   = (const float*)d_in[15];
    const float* t_w2    = (const float*)d_in[16];
    const float* t_b2    = (const float*)d_in[17];
    float* out = (float*)d_out;

    float* ws   = (float*)d_ws;
    float* h_r  = ws;
    float* h_t  = h_r + (size_t)BB * 64 * 289;
    float* r_l  = h_t + (size_t)BB * 64 * 289;
    float* t_l  = r_l + (size_t)BB * 1024 * 289;

    {
        int total = BB * 64 * 289;
        int blocks = (total + 255) / 256;
        conv3x3_17_kernel<<<blocks, 256, 0, stream>>>(
            p_low, r_w1, r_b1, r_gamma, r_beta, r_mean, r_var, h_r, 64, 64, 1, total);
        conv3x3_17_kernel<<<blocks, 256, 0, stream>>>(
            p_low, t_w1, t_b1, t_gamma, t_beta, t_mean, t_var, h_t, 64, 64, 1, total);
    }
    {
        int totalR = BB * 1024 * 289;
        conv3x3_17_kernel<<<(totalR + 255) / 256, 256, 0, stream>>>(
            h_r, r_w2, r_b2, r_gamma, r_beta, r_mean, r_var, r_l, 64, 1024, 0, totalR);
        int totalT = BB * 48 * 289;
        conv3x3_17_kernel<<<(totalT + 255) / 256, 256, 0, stream>>>(
            h_t, t_w2, t_b2, t_gamma, t_beta, t_mean, t_var, t_l, 64, 48, 0, totalT);
    }
    local_rct_attn_kernel<<<dim3(256, BB), 256, 0, stream>>>(feature, r_l, t_l, out);
}
